// DetectionLoss_91328184582378
// MI455X (gfx1250) — compile-verified, matrix operand order checked by reading, not run on hardware
//
#include <hip/hip_runtime.h>
#include <hip/hip_bf16.h>

typedef __attribute__((ext_vector_type(16))) _Float16 v16h;
typedef __attribute__((ext_vector_type(8)))  float    v8f;

typedef __attribute__((address_space(1))) int* gptr_i32;
typedef __attribute__((address_space(3))) int* lptr_i32;

#define NB   32          // batch
#define NM   20          // boxes per image
#define NTOT 64512       // anchors per image across levels (49152+12288+3072)

#if defined(__AMDGCN__) && __has_builtin(__builtin_amdgcn_global_load_async_to_lds_b32) && \
    __has_builtin(__builtin_amdgcn_s_wait_asynccnt)
#define USE_ASYNC_LDS 1
#else
#define USE_ASYNC_LDS 0
#endif

// -------- ws layout (bytes) --------
// 0      : blkpart  float[2688*3]   (per-block partials: loc, obj_pos, cls)
// 32768  : npos     int[96]
// 33280  : nneg     int[96]
// 33792  : objneg   float[96]
// 65536  : keys     u64[32*64512]   (~16.5 MB)

__global__ void dl_init(int* __restrict__ npos, int* __restrict__ nneg) {
  int t = blockIdx.x * blockDim.x + threadIdx.x;
  if (t < 96) { npos[t] = 0; nneg[t] = 0; }
}

// One thread per (h,w) pixel; handles A=3 anchors. blockDim = 256.
__global__ void dl_passA(const float* __restrict__ pred,
                         const float* __restrict__ boxes,
                         const int*   __restrict__ labels,
                         unsigned long long* __restrict__ keys,
                         float* __restrict__ blkpart,
                         int* __restrict__ npos, int* __restrict__ nneg,
                         int H, int W, float strd, int level_base,
                         int level_idx, int blk_off) {
  const int HW  = H * W;
  const int bpb = HW / 256;                  // blocks per batch image
  const int batch = blockIdx.x / bpb;
  const int pix   = (blockIdx.x % bpb) * 256 + threadIdx.x;
  const int t = threadIdx.x;

  __shared__ float sbox[NM * 4];
  __shared__ int   slab[NM];
  __shared__ float swl[8], swo[8], swc[8];
  __shared__ int   swp[8], swn[8];

#if USE_ASYNC_LDS
  // Stage boxes/labels into LDS via the CDNA5 async-to-LDS path (ASYNCcnt).
  if (t < NM * 4)
    __builtin_amdgcn_global_load_async_to_lds_b32(
        (gptr_i32)(int*)(void*)(boxes + batch * NM * 4 + t),
        (lptr_i32)(int*)&sbox[t], 0, 0);
  if (t < NM)
    __builtin_amdgcn_global_load_async_to_lds_b32(
        (gptr_i32)(int*)(void*)(labels + batch * NM + t),
        (lptr_i32)&slab[t], 0, 0);
  __builtin_amdgcn_s_wait_asynccnt(0);
#else
  if (t < NM * 4) sbox[t] = boxes[batch * NM * 4 + t];
  if (t < NM)     slab[t] = labels[batch * NM + t];
#endif
  __syncthreads();

  const int w = pix % W, h = pix / W;
  const float cx = ((float)w + 0.5f) * strd;
  const float cy = ((float)h + 0.5f) * strd;

  float loc = 0.f, objp = 0.f, clss = 0.f;
  int cp = 0, cn = 0;

  unsigned long long* kout =
      keys + (size_t)batch * NTOT + level_base + (size_t)pix * 3;
  const float* pbase = pred + (size_t)batch * 24 * HW + pix;

  for (int a = 0; a < 3; ++a) {
    const float sz   = (float)(a + 2) * strd;      // sizes 2,3,4 * stride
    const float half = 0.5f * sz;
    const float ax0 = cx - half, ay0 = cy - half;
    const float ax1 = cx + half, ay1 = cy + half;
    const float area_a = sz * sz;

    float best = -1.f; int bidx = 0;
    for (int m = 0; m < NM; ++m) {
      float bx0 = sbox[4*m], by0 = sbox[4*m+1];
      float bx1 = sbox[4*m+2], by1 = sbox[4*m+3];
      float lx = fmaxf(ax0, bx0), ly = fmaxf(ay0, by0);
      float rx = fminf(ax1, bx1), ry = fminf(ay1, by1);
      float iw = fmaxf(rx - lx, 0.f), ih = fmaxf(ry - ly, 0.f);
      float inter = iw * ih;
      float ab = (bx1 - bx0) * (by1 - by0);
      float iou = inter / fmaxf(area_a + ab - inter, 1e-9f);
      iou *= (slab[m] > 0) ? 1.f : 0.f;
      if (iou > best) { best = iou; bidx = m; }   // strict > == first-max (argmax)
    }
    const bool pos = (best >= 0.5f);
    const bool neg = (best < 0.3f);

    const float* pc = pbase + (size_t)(a * 8) * HW;
    const float x = pc[(size_t)4 * HW];           // objectness logit (always read)
    const float bce = fmaxf(x, 0.f) - (pos ? x : 0.f) + log1pf(expf(-fabsf(x)));

    if (pos) {
      objp += bce; cp++;
      float d0 = pc[0],              d1 = pc[HW];
      float d2 = pc[(size_t)2 * HW], d3 = pc[(size_t)3 * HW];
      float l0 = pc[(size_t)5 * HW], l1 = pc[(size_t)6 * HW], l2 = pc[(size_t)7 * HW];
      float mb0 = sbox[4*bidx], mb1 = sbox[4*bidx+1];
      float mb2 = sbox[4*bidx+2], mb3 = sbox[4*bidx+3];
      float s = 0.f, d;
      d = fabsf(d0 - mb0); s += (d < 1.f) ? 0.5f * d * d : (d - 0.5f);
      d = fabsf(d1 - mb1); s += (d < 1.f) ? 0.5f * d * d : (d - 0.5f);
      d = fabsf(d2 - mb2); s += (d < 1.f) ? 0.5f * d * d : (d - 0.5f);
      d = fabsf(d3 - mb3); s += (d < 1.f) ? 0.5f * d * d : (d - 0.5f);
      loc += 0.25f * s;                           // mean over 4 coords
      float mx  = fmaxf(l0, fmaxf(l1, l2));
      float lse = mx + logf(expf(l0 - mx) + expf(l1 - mx) + expf(l2 - mx));
      int tgt = slab[bidx] - 1; tgt = tgt < 0 ? 0 : (tgt > 2 ? 2 : tgt);
      float lt = (tgt == 0) ? l0 : ((tgt == 1) ? l1 : l2);
      clss += lse - lt;
    }

    unsigned long long key = 0ULL;
    if (neg) {
      cn++;
      // bce >= 0 -> float bits are monotone; embed index for stable tie-break
      key = ((unsigned long long)(__float_as_uint(bce) | 0x80000000u) << 32)
          | (unsigned long long)(0x7FFFFFFFu - (unsigned)(pix * 3 + a));
    }
    kout[a] = key;
  }

  // Pack counts early so the integer temps die before the shuffle reduction.
  const float packed = (float)(cp + (cn << 8));

  // ---- wave reduction: floats via shuffles ----
  for (int off = 16; off; off >>= 1) {
    loc  += __shfl_xor(loc,  off, 32);
    objp += __shfl_xor(objp, off, 32);
    clss += __shfl_xor(clss, off, 32);
  }

  // ---- wave reduction of packed counts via one WMMA (v_wmma_f32_16x16x32_f16)
  // A[m][0] = packed count of lane m (m<16); A[m][8] = packed count of lane m+16
  // (16-bit A layout: a[0] = K0 for lanes 0-15, K8 for lanes 16-31). B = ones ->
  // D[m][n] = rowsum = packed[m] + packed[m+16]. D rows 0..7 live in lanes<16,
  // rows 8..15 in lanes>=16, so sum(c[0..7]) + shfl_xor(.,16) = total.
  // packed <= 3 + 3*256 = 771 (exact in f16); wave total <= 24672 (exact in f32).
  {
    v16h av = {};
    av[0] = (_Float16)packed;
    const _Float16 one = (_Float16)1.0f;
    v16h bv = {one, one, one, one, one, one, one, one,
               one, one, one, one, one, one, one, one};
    v8f cacc = {};
    cacc = __builtin_amdgcn_wmma_f32_16x16x32_f16(
        false, av, false, bv, (short)0, cacc, false, false);
    float srow = cacc[0] + cacc[1] + cacc[2] + cacc[3]
               + cacc[4] + cacc[5] + cacc[6] + cacc[7];
    float tot = srow + __shfl_xor(srow, 16, 32);
    int ti = (int)(tot + 0.5f);
    cp = ti & 255;
    cn = ti >> 8;
  }

  const int wave = t >> 5, lane = t & 31;
  if (lane == 0) { swl[wave] = loc; swo[wave] = objp; swc[wave] = clss;
                   swp[wave] = cp;  swn[wave] = cn; }
  __syncthreads();
  if (t == 0) {
    float L = 0, O = 0, Cc = 0; int P = 0, Nn = 0;
    for (int i = 0; i < 8; ++i) { L += swl[i]; O += swo[i]; Cc += swc[i];
                                  P += swp[i]; Nn += swn[i]; }
    int gblk = blk_off + blockIdx.x;
    blkpart[3 * gblk + 0] = L;
    blkpart[3 * gblk + 1] = O;
    blkpart[3 * gblk + 2] = Cc;
    int prob = batch * 3 + level_idx;
    atomicAdd(&npos[prob], P);
    atomicAdd(&nneg[prob], Nn);
  }
}

// One block per (batch, level) problem: exact radix-select of k-th largest key,
// then sum bce over selected negatives.
__global__ void dl_select(const unsigned long long* __restrict__ keys,
                          const int* __restrict__ npos,
                          const int* __restrict__ nneg,
                          float* __restrict__ objneg) {
  const int p = blockIdx.x;
  const int b = p / 3, lvl = p % 3;
  const int base = (lvl == 0) ? 0 : ((lvl == 1) ? 49152 : 61440);
  const int N    = (lvl == 0) ? 49152 : ((lvl == 1) ? 12288 : 3072);
  const unsigned long long* K = keys + (size_t)b * NTOT + base;

  const int np = npos[p], nn = nneg[p];
  const int k = (np == 0) ? min(100, nn) : min(3 * np, nn);

  __shared__ unsigned hist[256];
  __shared__ unsigned long long sh_pref;
  __shared__ int sh_krem;
  __shared__ float sred[256];
  if (threadIdx.x == 0) { sh_pref = 0ULL; sh_krem = k; }
  __syncthreads();

  if (k > 0) {
    for (int pass = 0; pass < 8; ++pass) {
      const int shift = 56 - 8 * pass;
      hist[threadIdx.x] = 0;
      __syncthreads();
      const unsigned long long pref = sh_pref;
      for (int i = threadIdx.x; i < N; i += blockDim.x) {
        unsigned long long key = K[i];
        bool match = (pass == 0) ||
                     ((key >> (shift + 8)) == (pref >> (shift + 8)));
        if (match) atomicAdd(&hist[(unsigned)(key >> shift) & 255u], 1u);
      }
      __syncthreads();
      if (threadIdx.x == 0) {
        int krem = sh_krem;
        unsigned cum = 0;
        int d = 255;
        for (; d >= 0; --d) {
          unsigned c = hist[d];
          if (cum + c >= (unsigned)krem) break;
          cum += c;
        }
        if (d < 0) d = 0;  // unreachable: matched count >= krem invariant
        sh_krem = krem - (int)cum;
        sh_pref = pref | ((unsigned long long)(unsigned)d << shift);
      }
      __syncthreads();
    }
  }

  // Sum bce over the exactly-k keys >= threshold (all keys distinct).
  const unsigned long long T = sh_pref;
  float s = 0.f;
  if (k > 0) {
    for (int i = threadIdx.x; i < N; i += blockDim.x) {
      unsigned long long key = K[i];
      if ((key >> 63) && key >= T)
        s += __uint_as_float((unsigned)(key >> 32) & 0x7FFFFFFFu);
    }
  }
  sred[threadIdx.x] = s;
  __syncthreads();
  for (int st = 128; st; st >>= 1) {
    if (threadIdx.x < st) sred[threadIdx.x] += sred[threadIdx.x + st];
    __syncthreads();
  }
  if (threadIdx.x == 0) objneg[p] = sred[0];
}

__global__ void dl_combine(const float* __restrict__ blkpart,
                           const int* __restrict__ npos,
                           const int* __restrict__ nneg,
                           const float* __restrict__ objneg,
                           float* __restrict__ out) {
  __shared__ float sl[256], so[256], sc[256];
  const int t = threadIdx.x;
  float l = 0, o = 0, c = 0;
  for (int i = t; i < 2688; i += 256) {
    l += blkpart[3 * i + 0];
    o += blkpart[3 * i + 1];
    c += blkpart[3 * i + 2];
  }
  sl[t] = l; so[t] = o; sc[t] = c;
  __syncthreads();
  for (int st = 128; st; st >>= 1) {
    if (t < st) { sl[t] += sl[t + st]; so[t] += so[t + st]; sc[t] += sc[t + st]; }
    __syncthreads();
  }
  if (t == 0) {
    long long tp = 0, ts = 0;
    float on = 0.f;
    for (int p = 0; p < 96; ++p) {
      int np = npos[p], nn = nneg[p];
      int k = (np == 0) ? min(100, nn) : min(3 * np, nn);
      tp += np; ts += k; on += objneg[p];
    }
    float loc = sl[0], obj = so[0] + on, cls = sc[0];
    long long nrm = tp + ts; if (nrm < 1) nrm = 1;
    long long npd = tp;      if (npd < 1) npd = 1;
    float lloc = 2.0f * loc / (float)nrm;
    float lobj = obj / (float)nrm;
    float lcls = cls / (float)npd;
    out[0] = lloc + lobj + lcls;
    out[1] = lloc;
    out[2] = lobj;
    out[3] = lcls;
  }
}

extern "C" void kernel_launch(void* const* d_in, const int* in_sizes, int n_in,
                              void* d_out, int out_size, void* d_ws, size_t ws_size,
                              hipStream_t stream) {
  (void)in_sizes; (void)n_in; (void)out_size; (void)ws_size;
  const float* p1     = (const float*)d_in[0];
  const float* p2     = (const float*)d_in[1];
  const float* p3     = (const float*)d_in[2];
  // d_in[3..5] = anchors: recomputed exactly on-device, unused
  const float* boxes  = (const float*)d_in[6];
  const int*   labels = (const int*)d_in[7];
  float* out = (float*)d_out;

  char* ws = (char*)d_ws;
  float* blkpart = (float*)ws;                              // 32256 B
  int*   npos    = (int*)(ws + 32768);                      // 384 B
  int*   nneg    = (int*)(ws + 33280);                      // 384 B
  float* objneg  = (float*)(ws + 33792);                    // 384 B
  unsigned long long* keys = (unsigned long long*)(ws + 65536); // 16.5 MB

  dl_init<<<1, 128, 0, stream>>>(npos, nneg);
  dl_passA<<<2048, 256, 0, stream>>>(p1, boxes, labels, keys, blkpart, npos, nneg,
                                     128, 128, 8.0f, 0,     0, 0);
  dl_passA<<<512,  256, 0, stream>>>(p2, boxes, labels, keys, blkpart, npos, nneg,
                                     64,  64,  16.0f, 49152, 1, 2048);
  dl_passA<<<128,  256, 0, stream>>>(p3, boxes, labels, keys, blkpart, npos, nneg,
                                     32,  32,  32.0f, 61440, 2, 2560);
  dl_select<<<96, 256, 0, stream>>>(keys, npos, nneg, objneg);
  dl_combine<<<1, 256, 0, stream>>>(blkpart, npos, nneg, objneg, out);
}
